// TreeRNN_86380382257426
// MI455X (gfx1250) — matrix operand order repaired and checked by reading, not compiled, not run on hardware
//
#include <hip/hip_runtime.h>
#include <hip/hip_bf16.h>
#include <math.h>

typedef __attribute__((ext_vector_type(16))) _Float16 v16h;
typedef __attribute__((ext_vector_type(8)))  float    v8f;

// Tree constants (complete 8-ary tree, depth 7)
#define NTOT 299593

// ---------------------------------------------------------------------------
// Pack a 128x128 f32 row-major weight (K x N) into WMMA B-fragment layout:
// packed[(kb*8 + nt)*512 + lane*16 + e]  with
//   k = kb*32 + (lane>>4)*8 + (e&7) + (e>>3)*16 ,  n = nt*16 + (lane&15)
// (B 32x16 f16 layout: lanes 0-15 hold K=0..7,16..23, lanes 16-31 hold K=8..15,24..31)
// ---------------------------------------------------------------------------
__global__ void pack_weights_kernel(const float* __restrict__ Win,
                                    const float* __restrict__ U,
                                    const float* __restrict__ Wout,
                                    _Float16* __restrict__ packed) {
    int gid = blockIdx.x * blockDim.x + threadIdx.x;   // 0 .. 3*16384-1
    if (gid >= 3 * 16384) return;
    int w   = gid >> 14;
    int fid = gid & 16383;
    int e   = fid & 15;
    int L   = (fid >> 4) & 31;
    int nt  = (fid >> 9) & 7;
    int kb  = fid >> 12;
    int n   = nt * 16 + (L & 15);
    int k   = kb * 32 + ((L >> 4) << 3) + (e & 7) + ((e >> 3) << 4);
    const float* src = (w == 0) ? Win : (w == 1) ? U : Wout;
    packed[gid] = (_Float16)src[k * 128 + n];
}

// ---------------------------------------------------------------------------
// Leaf level: h = tanh(h_input), stored as f16
// ---------------------------------------------------------------------------
__global__ void leaf_tanh_kernel(const float* __restrict__ h_input,
                                 _Float16* __restrict__ h16,
                                 int offElems, int totalElems) {
    int i = blockIdx.x * blockDim.x + threadIdx.x;
    if (i < totalElems)
        h16[offElems + i] = (_Float16)tanhf(h_input[offElems + i]);
}

// ---------------------------------------------------------------------------
// Unified WMMA GEMM kernel: one block = 16 rows x 128 cols, 8 waves,
// each wave computes one 16x16 tile via v_wmma_f32_16x16x32_f16 (K loop = 4).
//   mode 0: A = emb[x*mask] gather (f32->f16);  D = (A@W_in + b_in)*mask -> h_input (f32)
//   mode 1: A = sum of 8 contiguous children of h16; D = tanh(h_input + A@U) -> h16
//   mode 2: A = h16;                            D = A@W_out + b_out -> out (f32)
// ---------------------------------------------------------------------------
__global__ void treernn_gemm_kernel(int mode, int rows, int rowStart,
                                    const int*   __restrict__ x,
                                    const int*   __restrict__ mask,
                                    const float* __restrict__ emb,
                                    float*       __restrict__ h_input,
                                    _Float16*    __restrict__ h16,
                                    const _Float16* __restrict__ packedB,
                                    const float* __restrict__ bias,
                                    float*       __restrict__ out) {
    // A staged in LDS already in WMMA A-fragment layout: [kb][lane][elem]
    __shared__ __align__(32) _Float16 aPack[4][32][16];   // 4 KB

    const int tid      = threadIdx.x;
    const int blockRow = blockIdx.x * 16;

    // ---- build 16x128 A tile (f16, fragment-packed) ----
#pragma unroll
    for (int j = 0; j < 8; ++j) {
        int idx = j * 256 + tid;          // 0..2047 ; consecutive lanes -> consecutive k
        int m   = idx >> 7;               // 0..15
        int k   = idx & 127;              // 0..127
        int gm  = blockRow + m;
        float val = 0.f;
        if (gm < rows) {
            if (mode == 0) {
                int xi = x[gm] * mask[gm];
                val = emb[xi * 128 + k];
            } else if (mode == 1) {
                int node  = rowStart + gm;
                int cbase = (node * 8 + 1) * 128 + k;   // children 8i+1..8i+8
                float s = 0.f;
#pragma unroll
                for (int c = 0; c < 8; ++c) s += (float)h16[cbase + c * 128];
                val = s;
            } else {
                val = (float)h16[gm * 128 + k];
            }
        }
        // A 16x32 f16 fragment mapping (per 32-wide K block):
        //   kk in [0,8):lane=m,e=kk  [8,16):lane=m+16,e=kk-8
        //   [16,24):lane=m,e=kk-8    [24,32):lane=m+16,e=kk-16
        int kb   = k >> 5, kk = k & 31;
        int lane = m + (((kk >> 3) & 1) << 4);
        int elem = (kk & 7) + ((kk >> 4) << 3);
        aPack[kb][lane][elem] = (_Float16)val;
    }
    __syncthreads();

    // ---- WMMA: each wave owns N-tile `wave` (cols 16*wave .. 16*wave+15) ----
    const int wave = tid >> 5;
    const int lane = tid & 31;

    v8f acc = {};
#pragma unroll
    for (int kb = 0; kb < 4; ++kb) {
        v16h a = *(const v16h*)(&aPack[kb][lane][0]);
        v16h b = *(const v16h*)(packedB + (size_t)(kb * 8 + wave) * 512 + lane * 16);
        acc = __builtin_amdgcn_wmma_f32_16x16x32_f16(
            /*neg_a=*/false, a, /*neg_b=*/false, b,
            /*c_mod=*/(short)0, acc, /*reuse_a=*/false, /*reuse_b=*/false);
    }

    // ---- epilogue: C/D layout = lanes 0-15 rows 0-7, lanes 16-31 rows 8-15 ----
    const int col   = wave * 16 + (lane & 15);
    const int rbase = (lane >> 4) << 3;
#pragma unroll
    for (int r = 0; r < 8; ++r) {
        int gm = blockRow + rbase + r;
        if (gm >= rows) continue;
        float v = acc[r];
        if (mode == 0) {
            float mf = (float)mask[gm];
            h_input[gm * 128 + col] = (v + bias[col]) * mf;
        } else if (mode == 1) {
            int node = rowStart + gm;
            h16[node * 128 + col] =
                (_Float16)tanhf(h_input[node * 128 + col] + v);
        } else {
            out[gm * 128 + col] = v + bias[col];
        }
    }
}

// ---------------------------------------------------------------------------
extern "C" void kernel_launch(void* const* d_in, const int* in_sizes, int n_in,
                              void* d_out, int out_size, void* d_ws, size_t ws_size,
                              hipStream_t stream) {
    const int*   x     = (const int*)d_in[0];
    const int*   mask  = (const int*)d_in[1];
    /* d_in[2] = children (unused; children of i are 8i+1..8i+8) */
    const float* emb   = (const float*)d_in[3];
    const float* W_in  = (const float*)d_in[4];
    const float* b_in  = (const float*)d_in[5];
    const float* U     = (const float*)d_in[6];
    const float* W_out = (const float*)d_in[7];
    const float* b_out = (const float*)d_in[8];
    float*       out   = (float*)d_out;

    static const int LEVEL_OFFS[8] = {0, 1, 9, 73, 585, 4681, 37449, 299593};

    // Workspace layout
    char*  ws          = (char*)d_ws;
    size_t hinputBytes = (size_t)NTOT * 128 * sizeof(float);     // ~153 MB
    size_t h16Bytes    = (size_t)NTOT * 128 * sizeof(_Float16);  // ~77 MB
    float*    h_input  = (float*)ws;
    _Float16* h16      = (_Float16*)(ws + hinputBytes);
    _Float16* packW    = (_Float16*)(ws + hinputBytes + h16Bytes);
    _Float16* WinP     = packW;
    _Float16* UP       = packW + 16384;
    _Float16* WoutP    = packW + 32768;

    // 1) pack weights into WMMA B-fragment layout (f16)
    pack_weights_kernel<<<(3 * 16384 + 255) / 256, 256, 0, stream>>>(
        W_in, U, W_out, packW);

    // 2) h_input = (emb[x*mask] @ W_in + b_in) * mask
    int blocksN = (NTOT + 15) / 16;
    treernn_gemm_kernel<<<blocksN, 256, 0, stream>>>(
        0, NTOT, 0, x, mask, emb, h_input, h16, WinP, b_in, out);

    // 3) leaf level: h = tanh(h_input)
    int leafStart = LEVEL_OFFS[6];
    int leafElems = (NTOT - leafStart) * 128;
    leaf_tanh_kernel<<<(leafElems + 255) / 256, 256, 0, stream>>>(
        h_input, h16, leafStart * 128, leafElems);

    // 4) internal levels bottom-up: h = tanh(h_input + (sum children) @ U)
    for (int l = 5; l >= 0; --l) {
        int s = LEVEL_OFFS[l], e = LEVEL_OFFS[l + 1];
        int rowsL = e - s;
        treernn_gemm_kernel<<<(rowsL + 15) / 16, 256, 0, stream>>>(
            1, rowsL, s, x, mask, emb, h_input, h16, UP, b_in, out);
    }

    // 5) out = h @ W_out + b_out
    treernn_gemm_kernel<<<blocksN, 256, 0, stream>>>(
        2, NTOT, 0, x, mask, emb, h_input, h16, WoutP, b_out, out);
}
